// Adaptive_Fusion_44272522887899
// MI455X (gfx1250) — matrix operand order repaired and checked by reading, not compiled, not run on hardware
//
#include <hip/hip_runtime.h>
#include <stdint.h>

// ---------------------------------------------------------------------------
// Problem constants (reference: B=16, T=64, N=128, F=512, 8 heads x 64)
// ---------------------------------------------------------------------------
#define HEADS 8
#define DIMS  64
#define FEAT  512
#define BATCH 16
#define NN    128
#define TT    64
#define NTOK  (NN * TT)           // 8192 tokens per batch
#define EPS_LN 1e-5f
#define NEGV  (-32767.0f)         // mask value BEFORE the /sqrt(d) scaling
#define SOFT_SCALE 0.125f         // 1/sqrt(64)

typedef __attribute__((ext_vector_type(16))) __bf16          bf16x16;
typedef __attribute__((ext_vector_type(8)))  __bf16          bf16x8;
typedef __attribute__((ext_vector_type(8)))  float           f32x8;
typedef __attribute__((ext_vector_type(8)))  unsigned short  u16x8;

// ---------------------------------------------------------------------------
// helpers
// ---------------------------------------------------------------------------
__device__ __forceinline__ unsigned short f2bf(float f) {
  unsigned u = __builtin_bit_cast(unsigned, f);
  u += 0x7FFFu + ((u >> 16) & 1u);               // round-to-nearest-even
  return (unsigned short)(u >> 16);
}
__device__ __forceinline__ float bf2f(unsigned short h) {
  return __builtin_bit_cast(float, ((unsigned)h) << 16);
}
__device__ __forceinline__ u16x8 ld8(const unsigned short* p) {
  return *(const u16x8*)p;                        // 16B vector load
}
__device__ __forceinline__ bf16x8 ld8b(const unsigned short* p) {
  return *(const bf16x8*)p;                       // 16B vector load as bf16x8
}
// concatenate two 16B chunks into one WMMA operand; lowers to register
// coalescing (no per-halfword v_mov_b16 like a union copy would).
__device__ __forceinline__ bf16x16 mkfrag(bf16x8 lo, bf16x8 hi) {
  return __builtin_shufflevector(lo, hi, 0, 1, 2, 3, 4, 5, 6, 7,
                                         8, 9, 10, 11, 12, 13, 14, 15);
}
__device__ __forceinline__ f32x8 wmma_bf16(bf16x16 a, bf16x16 b, f32x8 c) {
  // D = A(16x32) x B(32x16) + C, f32 accumulate
  return __builtin_amdgcn_wmma_f32_16x16x32_bf16(false, a, false, b,
                                                 (short)0, c, false, false);
}

// ---------------------------------------------------------------------------
// Kernel 1: weight prep.  W (fin,fout) f32 row-major -> WT (fout,fin) bf16.
// ---------------------------------------------------------------------------
__global__ __launch_bounds__(256) void wprep(const float* __restrict__ W,
                                             unsigned short* __restrict__ WT) {
  int idx = blockIdx.x * 256 + threadIdx.x;       // over FEAT*FEAT
  int fo = idx >> 9, fi = idx & 511;
  WT[idx] = f2bf(W[fi * FEAT + fo]);
}

// ---------------------------------------------------------------------------
// Kernel 2: input transpose.  X (B,F,N,T) f32 -> Y (B, N*T, F) bf16.
// 64(f) x 64(t) LDS tile per (b, n, fblk); both phases lane-coalesced.
// ---------------------------------------------------------------------------
__global__ __launch_bounds__(256) void xpose(const float* __restrict__ X,
                                             unsigned short* __restrict__ Y) {
  int n  = blockIdx.x % NN;
  int fb = blockIdx.x / NN;                       // 0..7
  int b  = blockIdx.y;
  __shared__ unsigned short tile[64][72];
  int tx = threadIdx.x & 63;
  int ty = threadIdx.x >> 6;                      // 0..3
#pragma unroll
  for (int i = 0; i < 16; ++i) {
    int f = ty * 16 + i;
    float v = X[(((size_t)b * FEAT + fb * 64 + f) * NN + n) * TT + tx];
    tile[f][tx] = f2bf(v);
  }
  __syncthreads();
#pragma unroll
  for (int i = 0; i < 16; ++i) {
    int t = ty * 16 + i;
    Y[((size_t)b * NTOK + n * 64 + t) * FEAT + fb * 64 + tx] = tile[tx][t];
  }
}

// ---------------------------------------------------------------------------
// Kernel 3: colgemm.  Y[b,nt,fo] = act( sum_fi WT[fo,fi]*X[b,nt,fi] + bias[fo] )
// WG = 256 thr = 8 waves arranged 4(fo) x 2(nt); WG tile 128fo x 128nt.
// Per wave: 2x4 = 8 WMMA accumulators, K-loop step 32.
// X tile (128nt x 32k bf16, 8KB) staged via global_load_async_to_lds_b128.
// A-fragments (WT rows) load as two 16B k-chunks per lane straight from global (L2-hot).
// ---------------------------------------------------------------------------
template <bool RELU>
__global__ __launch_bounds__(256) void colgemm(
    const unsigned short* __restrict__ X,    // [B][NTOK][FEAT] bf16
    const unsigned short* __restrict__ WT,   // [FEAT][FEAT]    bf16
    const float* __restrict__ bias,          // [FEAT]          f32
    unsigned short* __restrict__ Y) {        // [B][NTOK][FEAT] bf16
  __shared__ __attribute__((aligned(16))) unsigned short xlds[128 * 32];

  const int tid  = threadIdx.x;
  const int lane = tid & 31;
  const int wave = tid >> 5;
  const int lh = lane >> 4, ln = lane & 15;
  const int wm = wave & 3;                       // fo sub-block
  const int wn = wave >> 2;                      // nt sub-block
  const int b   = blockIdx.z;
  const int fo0 = blockIdx.y * 128 + wm * 32;
  const int ntB = blockIdx.x * 128;
  const size_t xbase = ((size_t)b * NTOK + ntB) * FEAT;

  f32x8 zero = {};
  f32x8 acc[2][4];
#pragma unroll
  for (int i = 0; i < 2; ++i)
#pragma unroll
    for (int j = 0; j < 4; ++j) acc[i][j] = zero;

  for (int k0 = 0; k0 < FEAT; k0 += 32) {
    // ---- async stage X tile: 128 rows x 32 k = 8KB = 512 x 16B chunks ----
#pragma unroll
    for (int c = 0; c < 2; ++c) {
      int q   = tid + 256 * c;                   // chunk id
      int row = q >> 2;
      int col = (q & 3) * 8;
      const unsigned short* gp = X + xbase + (size_t)row * FEAT + k0 + col;
      unsigned ldst = (unsigned)(uintptr_t)&xlds[q * 8];
      asm volatile("global_load_async_to_lds_b128 %0, %1, off"
                   :: "v"(ldst), "v"(gp) : "memory");
    }
    // prefetch next K-slab of the A operand
    __builtin_prefetch(WT + (size_t)(fo0 + ln) * FEAT + k0 + 32, 0, 0);
    asm volatile("s_wait_asynccnt 0x0" ::: "memory");
    __syncthreads();

    // ---- A fragments: WT rows, two 16B k-chunks (8*lh, 16+8*lh) per lane ----
    bf16x16 afrag[2];
#pragma unroll
    for (int mi = 0; mi < 2; ++mi) {
      const unsigned short* ar =
          WT + (size_t)(fo0 + mi * 16 + ln) * FEAT + k0 + lh * 8;
      afrag[mi] = mkfrag(ld8b(ar), ld8b(ar + 16));
    }
    // ---- B fragments from LDS: row nt, two 16B k-chunks (8*lh, 16+8*lh) per lane ----
    bf16x16 bfrag[4];
#pragma unroll
    for (int ni = 0; ni < 4; ++ni) {
      const unsigned short* br = &xlds[(wn * 64 + ni * 16 + ln) * 32 + lh * 8];
      bfrag[ni] = mkfrag(ld8b(br), ld8b(br + 16));
    }
#pragma unroll
    for (int mi = 0; mi < 2; ++mi)
#pragma unroll
      for (int ni = 0; ni < 4; ++ni)
        acc[mi][ni] = wmma_bf16(afrag[mi], bfrag[ni], acc[mi][ni]);
    __syncthreads();                             // protect LDS for next stage
  }

  // ---- epilogue: bias + relu + bf16 pack; C layout: M = r + 8*lh, N = ln --
#pragma unroll
  for (int mi = 0; mi < 2; ++mi) {
    float bv[8];
#pragma unroll
    for (int r = 0; r < 8; ++r) bv[r] = bias[fo0 + mi * 16 + r + 8 * lh];
#pragma unroll
    for (int ni = 0; ni < 4; ++ni) {
      int nt = ntB + wn * 64 + ni * 16 + ln;
      unsigned short* yr =
          Y + ((size_t)b * NTOK + nt) * FEAT + fo0 + mi * 16 + 8 * lh;
#pragma unroll
      for (int r = 0; r < 8; r += 2) {
        float v0 = acc[mi][ni][r] + bv[r];
        float v1 = acc[mi][ni][r + 1] + bv[r + 1];
        if (RELU) { v0 = fmaxf(v0, 0.f); v1 = fmaxf(v1, 0.f); }
        unsigned pk = (unsigned)f2bf(v0) | ((unsigned)f2bf(v1) << 16);
        *(unsigned*)(yr + r) = pk;
      }
    }
  }
}

// ---------------------------------------------------------------------------
// Kernel 4: attention.  One wave per (b, head, node): 64x64 causal block.
// S = Q K^T (bf16 WMMA, f32 acc), mask -> *1/8 -> softmax (shfl width-16,
// matching C-fragment row distribution), P round-trips LDS to re-fragment as
// an A operand, V^T staged in LDS, O = P V.
// ---------------------------------------------------------------------------
__global__ __launch_bounds__(64) void attn_kernel(
    const unsigned short* __restrict__ Q, const unsigned short* __restrict__ K,
    const unsigned short* __restrict__ V, unsigned short* __restrict__ O) {
  __shared__ __attribute__((aligned(16))) unsigned short smem[2 * 2 * 64 * 72];
  const int wave = threadIdx.x >> 5;
  const int lane = threadIdx.x & 31;
  const int lh = lane >> 4, ln = lane & 15;
  unsigned short* Pl = smem + wave * (2 * 64 * 72);  // 64 x 64, stride 72
  unsigned short* VT = Pl + 64 * 72;                 // V transposed (d, t2)

  int g = blockIdx.x * 2 + wave;                     // (b, h, n)
  int b  = g / (HEADS * NN);
  int r0 = g % (HEADS * NN);
  int h  = r0 / NN;
  int n  = r0 % NN;
  const size_t base = ((size_t)b * NTOK + (size_t)n * TT) * FEAT + h * DIMS;

  // Q A-fragments: row t1 = i*16 + ln, K(d) chunks per lane-half
  bf16x16 qa[4][2];
#pragma unroll
  for (int i = 0; i < 4; ++i)
#pragma unroll
    for (int kk = 0; kk < 2; ++kk) {
      const unsigned short* p =
          Q + base + (size_t)(i * 16 + ln) * FEAT + kk * 32 + lh * 8;
      qa[i][kk] = mkfrag(ld8b(p), ld8b(p + 16));
    }

  // S = Q K^T   (B-fragment column t2 = K row t2 -> contiguous in d)
  f32x8 zero = {};
  f32x8 s[4][4];
#pragma unroll
  for (int j = 0; j < 4; ++j) {
    bf16x16 kb[2];
#pragma unroll
    for (int kk = 0; kk < 2; ++kk) {
      const unsigned short* p =
          K + base + (size_t)(j * 16 + ln) * FEAT + kk * 32 + lh * 8;
      kb[kk] = mkfrag(ld8b(p), ld8b(p + 16));
    }
#pragma unroll
    for (int i = 0; i < 4; ++i) {
      f32x8 a = zero;
      a = wmma_bf16(qa[i][0], kb[0], a);
      a = wmma_bf16(qa[i][1], kb[1], a);
      s[i][j] = a;
    }
  }

  // causal mask (NEG before scaling, as in the reference), then scale
#pragma unroll
  for (int i = 0; i < 4; ++i)
#pragma unroll
    for (int j = 0; j < 4; ++j)
#pragma unroll
      for (int r = 0; r < 8; ++r) {
        int t1 = i * 16 + r + 8 * lh;
        int t2 = j * 16 + ln;
        float v = s[i][j][r];
        if (t2 > t1) v = NEGV;
        s[i][j][r] = v * SOFT_SCALE;
      }

  // softmax over t2 (64 = 4 j-tiles x 16 lanes of this half) + P -> LDS
#pragma unroll
  for (int i = 0; i < 4; ++i)
#pragma unroll
    for (int r = 0; r < 8; ++r) {
      float m = s[i][0][r];
#pragma unroll
      for (int j = 1; j < 4; ++j) m = fmaxf(m, s[i][j][r]);
#pragma unroll
      for (int off = 8; off >= 1; off >>= 1) m = fmaxf(m, __shfl_xor(m, off, 16));
      float p[4], sm = 0.f;
#pragma unroll
      for (int j = 0; j < 4; ++j) { p[j] = __expf(s[i][j][r] - m); sm += p[j]; }
#pragma unroll
      for (int off = 8; off >= 1; off >>= 1) sm += __shfl_xor(sm, off, 16);
      float inv = 1.f / sm;
      int t1 = i * 16 + r + 8 * lh;
#pragma unroll
      for (int j = 0; j < 4; ++j)
        Pl[t1 * 72 + j * 16 + ln] = f2bf(p[j] * inv);
    }

  // stage V^T (d, t2) into LDS: coalesced global reads, 2B transposed stores
  for (int t2 = 0; t2 < 64; ++t2) {
    unsigned v2 = *(const unsigned*)(V + base + (size_t)t2 * FEAT + 2 * lane);
    VT[(2 * lane) * 72 + t2]     = (unsigned short)(v2 & 0xFFFFu);
    VT[(2 * lane + 1) * 72 + t2] = (unsigned short)(v2 >> 16);
  }
  asm volatile("s_wait_dscnt 0x0" ::: "memory");   // LDS in-order per wave

  // P A-fragments from LDS
  bf16x16 pa[4][2];
#pragma unroll
  for (int i = 0; i < 4; ++i)
#pragma unroll
    for (int kk = 0; kk < 2; ++kk) {
      const unsigned short* p = &Pl[(i * 16 + ln) * 72 + kk * 32 + lh * 8];
      pa[i][kk] = mkfrag(ld8b(p), ld8b(p + 16));
    }
  // O = P V : B-fragment column d = VT row d -> contiguous in t2
#pragma unroll
  for (int j = 0; j < 4; ++j) {
    bf16x16 vb[2];
#pragma unroll
    for (int kk = 0; kk < 2; ++kk) {
      const unsigned short* p = &VT[(j * 16 + ln) * 72 + kk * 32 + lh * 8];
      vb[kk] = mkfrag(ld8b(p), ld8b(p + 16));
    }
#pragma unroll
    for (int i = 0; i < 4; ++i) {
      f32x8 a = zero;
      a = wmma_bf16(pa[i][0], vb[0], a);
      a = wmma_bf16(pa[i][1], vb[1], a);
#pragma unroll
      for (int r = 0; r < 8; ++r)
        O[base + (size_t)(i * 16 + r + 8 * lh) * FEAT + j * 16 + ln] = f2bf(a[r]);
    }
  }
}

// ---------------------------------------------------------------------------
// Kernel 5: residual add + LayerNorm over features.
// Thread per token. IN1 either f32 in (B,F,N,T) (original xl, coalesced
// across lanes) or bf16 token-row; OUT either bf16 token-row or f32 (B,F,N,T).
// ---------------------------------------------------------------------------
template <bool IN1_F32_BFNT, bool OUT_F32_BFNT>
__global__ __launch_bounds__(256) void add_ln(
    const void* __restrict__ in1, const unsigned short* __restrict__ in2,
    void* __restrict__ out) {
  int idx = blockIdx.x * 256 + threadIdx.x;       // over BATCH*NTOK tokens
  int b = idx / NTOK, nt = idx % NTOK;
  int n = nt / TT, t = nt % TT;
  const float* f1          = (const float*)in1;
  const unsigned short* h1 = (const unsigned short*)in1;
  const unsigned short* r2 = in2 + (size_t)idx * FEAT;

  float sum = 0.f, sq = 0.f;
  for (int c = 0; c < FEAT / 8; ++c) {
    u16x8 a2 = ld8(r2 + c * 8);
    u16x8 a1v = {};
    if (!IN1_F32_BFNT) a1v = ld8(h1 + (size_t)idx * FEAT + c * 8);
#pragma unroll
    for (int e = 0; e < 8; ++e) {
      int f = c * 8 + e;
      float x1 = IN1_F32_BFNT
                     ? f1[(((size_t)b * FEAT + f) * NN + n) * TT + t]
                     : bf2f(a1v[e]);
      float x = x1 + bf2f(a2[e]);
      sum += x; sq += x * x;
    }
  }
  float mean = sum * (1.f / FEAT);
  float var  = sq * (1.f / FEAT) - mean * mean;   // biased var, as jnp.var
  float inv  = rsqrtf(var + EPS_LN);

  for (int c = 0; c < FEAT / 8; ++c) {
    u16x8 a2 = ld8(r2 + c * 8);
    u16x8 a1v = {};
    if (!IN1_F32_BFNT) a1v = ld8(h1 + (size_t)idx * FEAT + c * 8);
#pragma unroll
    for (int e = 0; e < 8; ++e) {
      int f = c * 8 + e;
      float x1 = IN1_F32_BFNT
                     ? f1[(((size_t)b * FEAT + f) * NN + n) * TT + t]
                     : bf2f(a1v[e]);
      float z = (x1 + bf2f(a2[e]) - mean) * inv;
      if (OUT_F32_BFNT)
        ((float*)out)[(((size_t)b * FEAT + f) * NN + n) * TT + t] = z;
      else
        ((unsigned short*)out)[(size_t)idx * FEAT + f] = f2bf(z);
    }
  }
}

// ---------------------------------------------------------------------------
// host side
// ---------------------------------------------------------------------------
extern "C" void kernel_launch(void* const* d_in, const int* in_sizes, int n_in,
                              void* d_out, int out_size, void* d_ws,
                              size_t ws_size, hipStream_t stream) {
  (void)in_sizes; (void)n_in; (void)out_size; (void)ws_size;
  const float* xl = (const float*)d_in[0];
  const float* xh = (const float*)d_in[1];
  const float* Wm[6]   = {(const float*)d_in[2],  (const float*)d_in[4],
                          (const float*)d_in[6],  (const float*)d_in[8],
                          (const float*)d_in[10], (const float*)d_in[12]};
  const float* bq = (const float*)d_in[3],  *bk = (const float*)d_in[5];
  const float* bv = (const float*)d_in[7],  *bo = (const float*)d_in[9];
  const float* b1 = (const float*)d_in[11], *b2 = (const float*)d_in[13];

  // workspace layout: 6 x 512KB transposed bf16 weights, then 5 x 128MB
  // bf16 activation buffers (reused along the pipeline).
  const size_t WELEM = (size_t)FEAT * FEAT;                 // per weight
  const size_t AELEM = (size_t)BATCH * NTOK * FEAT;         // per activation
  unsigned short* wt[6];
  for (int i = 0; i < 6; ++i)
    wt[i] = (unsigned short*)((char*)d_ws + i * WELEM * 2);
  unsigned short* actb = (unsigned short*)((char*)d_ws + 6 * WELEM * 2);
  unsigned short* Xl = actb;                 // -> later: attention output
  unsigned short* Xh = Xl + AELEM;           // -> later: Wo projection output
  unsigned short* Qb = Xh + AELEM;           // -> later: y (post-LN1)
  unsigned short* Kb = Qb + AELEM;           // -> later: FFN hidden
  unsigned short* Vb = Kb + AELEM;           // -> later: FFN output

  // 1) weight transpose + bf16
  for (int i = 0; i < 6; ++i)
    wprep<<<(FEAT * FEAT) / 256, 256, 0, stream>>>(Wm[i], wt[i]);
  // 2) input transpose to (B, NT, F) bf16
  dim3 tgrid(NN * (FEAT / 64), BATCH);
  xpose<<<tgrid, 256, 0, stream>>>(xl, Xl);
  xpose<<<tgrid, 256, 0, stream>>>(xh, Xh);
  // 3-5) Q/K/V projections
  dim3 ggrid(NTOK / 128, FEAT / 128, BATCH);
  colgemm<false><<<ggrid, 256, 0, stream>>>(Xl, wt[0], bq, Qb);
  colgemm<true ><<<ggrid, 256, 0, stream>>>(Xh, wt[1], bk, Kb);
  colgemm<true ><<<ggrid, 256, 0, stream>>>(Xh, wt[2], bv, Vb);
  // 6) attention (writes into Xl slot, which is free now)
  attn_kernel<<<(BATCH * HEADS * NN) / 2, 64, 0, stream>>>(Qb, Kb, Vb, Xl);
  // 7) output projection (into Xh slot)
  colgemm<false><<<ggrid, 256, 0, stream>>>(Xl, wt[3], bo, Xh);
  // 8) y = LN(x_l + out)  (into Qb slot)
  add_ln<true, false><<<(BATCH * NTOK) / 256, 256, 0, stream>>>(xl, Xh, Qb);
  // 9-10) FFN
  colgemm<true ><<<ggrid, 256, 0, stream>>>(Qb, wt[4], b1, Kb);
  colgemm<false><<<ggrid, 256, 0, stream>>>(Kb, wt[5], b2, Vb);
  // 11) y2 = LN(ff + y) -> f32 (B,F,N,T) output
  add_ln<false, true><<<(BATCH * NTOK) / 256, 256, 0, stream>>>(Vb, Qb, d_out);
}
